// BottleneckBit_30116310680087
// MI455X (gfx1250) — matrix operand order repaired and checked by reading, not compiled, not run on hardware
//
#include <hip/hip_runtime.h>

typedef __attribute__((ext_vector_type(16))) __bf16        v16bf;
typedef __attribute__((ext_vector_type(8)))  float         v8f;
typedef __attribute__((ext_vector_type(4))) unsigned int   u32x4;
typedef __attribute__((ext_vector_type(8)))  int           v8i;
typedef __attribute__((ext_vector_type(4)))  int           v4i;

union Frag16 { v16bf v; u32x4 q[2]; unsigned short s[16]; };

__device__ __forceinline__ unsigned short f2bf(float f) {
  unsigned u = __float_as_uint(f);
  u += 0x7fffu + ((u >> 16) & 1u);          // round-to-nearest-even
  return (unsigned short)(u >> 16);
}

__device__ __forceinline__ unsigned lds_addr_of(const void* p) {
  // flat shared pointer: addr[31:0] is the LDS byte offset (aperture decode)
  return (unsigned)(unsigned long long)p;
}

// ---------------------------------------------------------------------------
// TDM: async 2D tile load (weights) Global -> LDS.
// Tile: 32 bf16 (X, contiguous) x 128 rows (Y, stride K elems).
// LDS pad: +4 DWORDs after every 16 DWORDs -> row stride 80 B (BKP = 40 bf16).
// Descriptor per CDNA5 ISA section 8.3/8.4 (D# group 0 + group 1, 2D tensor).
// This toolchain exposes the 6-arg builtin form:
//   (u32x4 g0, i32x8 g1, i32x4 g2, i32x4 g3, i32x8 gx, i32 cpol)
// ---------------------------------------------------------------------------
__device__ __forceinline__ void tdm_load_a(const unsigned short* gsrc,
                                           unsigned lds_byte, int K)
{
  unsigned long long ga = (unsigned long long)gsrc;
  u32x4 g0;
  g0[0] = 1u;                                   // count=1 (valid), user mode
  g0[1] = lds_byte;                             // lds_addr (bytes)
  g0[2] = (unsigned)ga;                         // global_addr[31:0]
  g0[3] = ((unsigned)(ga >> 32) & 0x01FFFFFFu)  // global_addr[56:32]
        | 0x80000000u;                          // type=2 ("image")
  v8i g1;
  g1[0] = (int)((1u << 16)    // data_size = 2 bytes
              | (1u << 20)    // pad_enable
              | (3u << 22)    // pad_interval: 16 DWORDs
              | (3u << 25));  // pad_amount:   4 DWORDs
  g1[1] = (int)(32u  << 16);  // tensor_dim0[15:0]  = 32  (== tile, in-bounds)
  g1[2] = (int)(128u << 16);  // tensor_dim1[15:0]  = 128
  g1[3] = (int)(32u  << 16);  // tile_dim0 = 32
  g1[4] = (int)128;           // tile_dim1 = 128, tile_dim2 = 0
  g1[5] = (int)K;             // tensor_dim0_stride[31:0] = K elems
  g1[6] = 0;                  // stride[47:32]=0, dim1_stride lo=0
  g1[7] = 0;
  v4i z4 = {0, 0, 0, 0};                         // groups 2/3 unused (2D)
  v8i z8 = {0, 0, 0, 0, 0, 0, 0, 0};             // extra group (clang-23 form)
  __builtin_amdgcn_tensor_load_to_lds(g0, g1, z4, z4, z8, 0);
}

// ---------------------------------------------------------------------------
// Kernel 1: per-output-channel ternary quantization + BN fold.
// Exact median of |w| via 32-step binary search over uint bit patterns.
// ---------------------------------------------------------------------------
#define QK_MAX 2304

__global__ __launch_bounds__(256)
void quant_fold_kernel(const float* __restrict__ w,  const float* __restrict__ cb,
                       const float* __restrict__ g,  const float* __restrict__ be,
                       const float* __restrict__ mu, const float* __restrict__ var,
                       unsigned short* __restrict__ wq, float* __restrict__ bfold,
                       int Krow, int Cin, int is3x3)
{
  __shared__ unsigned int ab[QK_MAX];
  __shared__ int cnt;
  const int co = blockIdx.x;
  const int t  = threadIdx.x;
  const float* row = w + (long)co * Krow;

  for (int j = t; j < Krow; j += 256) ab[j] = __float_as_uint(fabsf(row[j]));
  __syncthreads();

  unsigned sel[2];
  const int ks[2] = { Krow / 2 - 1, Krow / 2 };
  for (int p = 0; p < 2; ++p) {
    unsigned lo = 0u, hi = 0xFFFFFFFFu;
    while (lo < hi) {
      unsigned mid = lo + ((hi - lo) >> 1);
      if (t == 0) cnt = 0;
      __syncthreads();
      int c = 0;
      for (int j = t; j < Krow; j += 256) c += (ab[j] <= mid) ? 1 : 0;
      atomicAdd(&cnt, c);
      __syncthreads();
      int total = cnt;
      __syncthreads();
      if (total >= ks[p] + 1) hi = mid; else lo = mid + 1;
    }
    sel[p] = lo;
  }

  float med   = 0.5f * (__uint_as_float(sel[0]) + __uint_as_float(sel[1]));
  float s     = fmaxf(med, 1e-8f);
  float scale = g[co] / sqrtf(var[co] + 1e-5f);
  float inv_s = 1.0f / s;
  float wsc   = s * scale;                       // fold BN scale into weights

  for (int j = t; j < Krow; j += 256) {
    float q = rintf(row[j] * inv_s);
    q = fminf(1.0f, fmaxf(-1.0f, q));
    // 3x3: permute OIHW k = c*9 + tap  ->  tap-major k' = tap*Cin + c
    int jd = is3x3 ? ((j % 9) * Cin + (j / 9)) : j;
    wq[(long)co * Krow + jd] = f2bf(q * wsc);
  }
  if (t == 0) bfold[co] = cb[co] * scale + be[co] - mu[co] * scale;
}

// ---------------------------------------------------------------------------
// Kernel 2: NCHW f32 -> [C][N] bf16 (N = B*HW), coalesced along HW.
// ---------------------------------------------------------------------------
__global__ __launch_bounds__(256)
void xpose_kernel(const float* __restrict__ x, unsigned short* __restrict__ xT,
                  int C, int HW, long total, int N)
{
  long i = (long)blockIdx.x * 256 + threadIdx.x;
  if (i >= total) return;
  int  hw = (int)(i % HW);
  long r  = i / HW;
  int  c  = (int)(r % C);
  int  b  = (int)(r / C);
  xT[(long)c * N + (long)b * HW + hw] = f2bf(x[i]);
}

// ---------------------------------------------------------------------------
// Kernel 3: WMMA-tiled GEMM  D[M,N] = Wq[M,K] x Act[K->N] + bias, + epilogue.
// 128x128 block tile, BK=32, 8 wave32s each owning 64x32 (4x2 fragments).
// Double-buffered LDS; A tile streamed by the Tensor Data Mover (wave 0),
// B tile gathered on the vector path (transpose / implicit 3x3 im2col).
// actmode: 0 = none, 1 = SiLU, 2 = +residual(x NCHW) then SiLU.
// ---------------------------------------------------------------------------
#define BM 128
#define BN 128
#define BK 32
#define BKP 40   // padded LDS K-stride (bf16) -- matches TDM pad settings

__global__ __launch_bounds__(256)
void gemm_bit_kernel(const unsigned short* __restrict__ A,   // [M][K] bf16
                     const unsigned short* __restrict__ Bm,  // [Cin][N] bf16
                     const float* __restrict__ bias,         // [M]
                     const float* __restrict__ xres,         // NCHW f32 or null
                     unsigned short* __restrict__ outb,      // [M][N] bf16 or null
                     float* __restrict__ outf,               // NCHW f32 or null
                     int M, int N, int K, int HW, int conv3, int actmode)
{
  __shared__ unsigned short At[2][BM * BKP];
  __shared__ unsigned short Bt[2][BN * BKP];

  const int t    = threadIdx.x;
  const int lane = t & 31;
  const int wave = t >> 5;
  const int wm   = wave & 1;        // 2 wave rows of 64
  const int wn   = wave >> 1;       // 4 wave cols of 32
  const int m0   = blockIdx.y * BM;
  const int n0   = blockIdx.x * BN;
  const int l    = lane & 15;
  const int hi   = lane >> 4;

  v8f acc[4][2];
  #pragma unroll
  for (int a = 0; a < 4; ++a)
    #pragma unroll
    for (int b = 0; b < 2; ++b) acc[a][b] = (v8f)0.0f;

  const int brow = t >> 3, bseg = t & 7;   // B: 32 k-rows x 8 chunks of 16 n
  const int steps = K / BK;

  // ---- B tile: global gather into registers (issued early) ----
  auto gather_b = [&](int k0, Frag16& u) {
    if (!conv3) {
      const unsigned short* src = Bm + (long)(k0 + brow) * N + n0 + bseg * 16;
      u.q[0] = *(const u32x4*)(src);
      u.q[1] = *(const u32x4*)(src + 8);
    } else {
      const int tap = k0 >> 8;               // K-tile lies inside one tap
      const int dr  = tap / 3 - 1, dc = tap % 3 - 1;
      const int c   = (k0 & 255) + brow;
      #pragma unroll
      for (int i = 0; i < 16; ++i) {
        int n  = n0 + bseg * 16 + i;
        int b  = n / HW, hw = n - b * HW;
        int h  = hw / 14, w2 = hw - h * 14;
        int hs = h + dr,  ws3 = w2 + dc;
        unsigned short val = 0;
        if ((unsigned)hs < 14u && (unsigned)ws3 < 14u)
          val = Bm[(long)c * N + (long)b * HW + hs * 14 + ws3];
        u.s[i] = val;
      }
    }
  };
  // ---- B tile: registers -> LDS, stored transposed as [n][k] ----
  auto store_b = [&](const Frag16& u, unsigned short* Btile) {
    #pragma unroll
    for (int i = 0; i < 16; ++i)
      Btile[(bseg * 16 + i) * BKP + brow] = u.s[i];
  };

  // ---- prologue: stage tile 0 ----
  if (wave == 0)
    tdm_load_a(A + (long)m0 * K, lds_addr_of(&At[0][0]), K);
  {
    Frag16 u0;
    gather_b(0, u0);
    store_b(u0, Bt[0]);
  }

  int cur = 0;
  for (int it = 0; it < steps; ++it) {
    if (wave == 0) __builtin_amdgcn_s_wait_tensorcnt(0);  // A tile `it` landed
    __syncthreads();                                      // all LDS writes visible

    const bool have_next = (it + 1 < steps);
    Frag16 unext;
    if (have_next) {
      gather_b((it + 1) * BK, unext);                     // loads in flight...
      if (wave == 0)
        tdm_load_a(A + (long)m0 * K + (it + 1) * BK,
                   lds_addr_of(&At[cur ^ 1][0]), K);      // ...TDM in flight
    }

    // ---- compute on current buffers: 12 ds b128 reads + 8 WMMAs ----
    Frag16 af[4], bf[2];
    #pragma unroll
    for (int fm = 0; fm < 4; ++fm) {
      const unsigned short* rp = At[cur] + (wm * 64 + fm * 16 + l) * BKP;
      af[fm].q[0] = *(const u32x4*)(rp + hi * 8);         // K 0..7 / 8..15
      af[fm].q[1] = *(const u32x4*)(rp + 16 + hi * 8);    // K 16..23 / 24..31
    }
    #pragma unroll
    for (int fn = 0; fn < 2; ++fn) {
      const unsigned short* rp = Bt[cur] + (wn * 32 + fn * 16 + l) * BKP;
      bf[fn].q[0] = *(const u32x4*)(rp + hi * 8);         // K 0..7 / 8..15
      bf[fn].q[1] = *(const u32x4*)(rp + 16 + hi * 8);    // K 16..23 / 24..31
    }
    #pragma unroll
    for (int fm = 0; fm < 4; ++fm)
      #pragma unroll
      for (int fn = 0; fn < 2; ++fn)
        acc[fm][fn] = __builtin_amdgcn_wmma_f32_16x16x32_bf16(
            false, af[fm].v, false, bf[fn].v, (short)0, acc[fm][fn],
            false, false);

    if (have_next) store_b(unext, Bt[cur ^ 1]);           // fill next buffer
    cur ^= 1;
  }

  // ---- epilogue: bias (+residual) (+SiLU), write bf16 act or f32 NCHW ----
  #pragma unroll
  for (int fm = 0; fm < 4; ++fm) {
    #pragma unroll
    for (int fn = 0; fn < 2; ++fn) {
      const int nn   = n0 + wn * 32 + fn * 16 + l;
      const int bimg = nn / HW, hw = nn - bimg * HW;
      #pragma unroll
      for (int v = 0; v < 8; ++v) {
        const int m = m0 + wm * 64 + fm * 16 + hi * 8 + v;
        float val = acc[fm][fn][v] + bias[m];
        if (actmode == 2) val += xres[((long)bimg * M + m) * HW + hw];
        if (actmode != 0) val = val / (1.0f + __expf(-val));
        if (outf) outf[((long)bimg * M + m) * HW + hw] = val;
        else      outb[(long)m * N + nn] = f2bf(val);
      }
    }
  }
}

// ---------------------------------------------------------------------------
extern "C" void kernel_launch(void* const* d_in, const int* in_sizes, int n_in,
                              void* d_out, int out_size, void* d_ws, size_t ws_size,
                              hipStream_t stream)
{
  (void)in_sizes; (void)n_in; (void)out_size; (void)ws_size;
  const float* x   = (const float*)d_in[0];
  const float* w1  = (const float*)d_in[1];
  const float* b1  = (const float*)d_in[2];
  const float* g1  = (const float*)d_in[3];
  const float* be1 = (const float*)d_in[4];
  const float* m1  = (const float*)d_in[5];
  const float* v1  = (const float*)d_in[6];
  const float* w2  = (const float*)d_in[7];
  const float* b2  = (const float*)d_in[8];
  const float* g2  = (const float*)d_in[9];
  const float* be2 = (const float*)d_in[10];
  const float* m2  = (const float*)d_in[11];
  const float* v2  = (const float*)d_in[12];
  const float* w3  = (const float*)d_in[13];
  const float* b3  = (const float*)d_in[14];
  const float* g3  = (const float*)d_in[15];
  const float* be3 = (const float*)d_in[16];
  const float* m3  = (const float*)d_in[17];
  const float* v3  = (const float*)d_in[18];

  const int B = 64, Cin = 1024, HW = 196, Wd = 256, Cout = 1024;
  const int N = B * HW;   // 12544 = 98 * 128

  char* p = (char*)d_ws;
  auto alloc = [&](size_t bytes) {
    char* r = p; p += (bytes + 255) & ~(size_t)255; return r;
  };
  unsigned short* xT   = (unsigned short*)alloc((size_t)Cin  * N * 2);
  unsigned short* act1 = (unsigned short*)alloc((size_t)Wd   * N * 2);
  unsigned short* act2 = (unsigned short*)alloc((size_t)Wd   * N * 2);
  unsigned short* wq1  = (unsigned short*)alloc((size_t)Wd   * Cin * 2);
  unsigned short* wq2  = (unsigned short*)alloc((size_t)Wd   * Wd * 9 * 2);
  unsigned short* wq3  = (unsigned short*)alloc((size_t)Cout * Wd * 2);
  float* bf1 = (float*)alloc((size_t)Wd   * 4);
  float* bf2 = (float*)alloc((size_t)Wd   * 4);
  float* bf3 = (float*)alloc((size_t)Cout * 4);

  // quantize + BN-fold all three weight sets
  quant_fold_kernel<<<Wd,   256, 0, stream>>>(w1, b1, g1, be1, m1, v1, wq1, bf1, Cin,    Cin, 0);
  quant_fold_kernel<<<Wd,   256, 0, stream>>>(w2, b2, g2, be2, m2, v2, wq2, bf2, Wd * 9, Wd,  1);
  quant_fold_kernel<<<Cout, 256, 0, stream>>>(w3, b3, g3, be3, m3, v3, wq3, bf3, Wd,     Wd,  0);

  // x: NCHW f32 -> [C][N] bf16
  long total = (long)B * Cin * HW;
  xpose_kernel<<<(unsigned)((total + 255) / 256), 256, 0, stream>>>(x, xT, Cin, HW, total, N);

  // conv1 (1x1, K=1024) -> SiLU -> act1
  gemm_bit_kernel<<<dim3(N / BN, Wd / BM), 256, 0, stream>>>(
      wq1, xT, bf1, nullptr, act1, nullptr, Wd, N, Cin, HW, 0, 1);
  // conv2 (3x3 via implicit im2col, K=2304) -> SiLU -> act2
  gemm_bit_kernel<<<dim3(N / BN, Wd / BM), 256, 0, stream>>>(
      wq2, act1, bf2, nullptr, act2, nullptr, Wd, N, Wd * 9, HW, 1, 1);
  // conv3 (1x1, K=256) + residual + SiLU -> d_out (f32 NCHW)
  gemm_bit_kernel<<<dim3(N / BN, Cout / BM), 256, 0, stream>>>(
      wq3, act2, bf3, x, nullptr, (float*)d_out, Cout, N, Wd, HW, 0, 2);
}
